// MedMambaEncoderBlock_18717467476150
// MI455X (gfx1250) — compile-verified, matrix operand order checked by reading, not run on hardware
//
#include <hip/hip_runtime.h>

// ---------------------------------------------------------------------------
// MedMamba encoder block for gfx1250 (MI455X, wave32, WMMA).
// All GEMMs run through v_wmma_f32_16x16x32_bf16 (bf16 in, f32 accumulate).
// A-tile staging: register-pipelined global_load_b128 + v_cvt_pk_bf16_f32.
// B-tile staging: GLOBAL_LOAD_ASYNC_TO_LDS_B128 (ASYNCcnt), double-buffered,
// overlapped with the WMMA section; drained with s_wait_asynccnt.
// ---------------------------------------------------------------------------

typedef __attribute__((ext_vector_type(16))) __bf16 v16bf;
typedef __attribute__((ext_vector_type(8)))  __bf16 v8bf;
typedef __attribute__((ext_vector_type(4)))  __bf16 v4bf;
typedef __attribute__((ext_vector_type(8)))  float  v8f;
typedef __attribute__((ext_vector_type(4)))  float  v4f;

union BFrag { v16bf v; struct { v8bf lo, hi; } h; };

// ----------------------------- constants -----------------------------------
#define CB   32      // batch
#define CL   256     // seq len
#define CD   512     // model dim
#define CDI  1024    // d_inner
#define CDS  16      // d_state
#define CDTR 32      // dt_rank
#define CDFF 2048    // ffn dim
#define CPROJ 64     // dt_rank + 2*d_state

#define BM 128
#define BN 64
#define BK 32
#define LDS_A 40     // bf16 elems per row (80B = 5*16B -> 16B-aligned rows)
#define LDS_B 40

// ----------------------------- device math ---------------------------------
__device__ __forceinline__ float siluf(float x) {
    return x / (1.f + __expf(-x));
}
__device__ __forceinline__ float softplusf(float x) {
    return (x > 20.f) ? x : log1pf(__expf(x));
}
__device__ __forceinline__ float geluf(float x) {
    const float c = 0.7978845608028654f; // sqrt(2/pi)
    float x3 = x * x * x;
    return 0.5f * x * (1.f + tanhf(c * (x + 0.044715f * x3)));
}

// ----------------------------- WMMA GEMM ------------------------------------
// C[M,N] = epi( A[M,K](f32, lda) @ Wt[N,K](bf16)^T , bias, res )
// EPI: 0=none  1=bias+softplus  2=bias+gelu  3=+res  4=bias+res
// REQUIRES: M % 128 == 0, N % 64 == 0, K % 32 == 0 (true for every call here).
template <int EPI>
__global__ __launch_bounds__(256)
void k_gemm(const float* __restrict__ A, int lda, long long bsA,
            const __bf16* __restrict__ Wt, long long bsW,
            float* __restrict__ C, int ldc, long long bsC,
            const float* __restrict__ bias,
            const float* __restrict__ res, long long bsR,
            int M, int N, int K)
{
    __shared__ __align__(16) __bf16 As[BM * LDS_A];
    __shared__ __align__(16) __bf16 Bs[2 * BN * LDS_B];   // double-buffered

    const int tid  = threadIdx.x;
    const int lane = tid & 31;
    const int wave = tid >> 5;   // 0..7
    const int wm   = wave & 3;   // 4 waves along M (32 rows each)
    const int wn   = wave >> 2;  // 2 waves along N (32 cols each)
    const int bm0  = blockIdx.y * BM;
    const int bn0  = blockIdx.x * BN;
    const long long bz = blockIdx.z;

    A  += bz * bsA;
    Wt += bz * bsW;
    C  += bz * bsC;
    const float* resp = (EPI == 3 || EPI == 4) ? (res + bz * bsR) : nullptr;

    // B-tile decomposition: 64x32 bf16 = 256 chunks of 16B, one per thread.
    const int b_n  = (tid >> 2);        // n row 0..63
    const int b_c8 = (tid & 3) * 8;     // 8-bf16 (16B) column group

    v4f ra[4];

    // A tile is 128 rows x 32 cols = 1024 float4 chunks; 256 threads -> 4 each.
    auto gloadA = [&](int k0) {
#pragma unroll
        for (int i = 0; i < 4; ++i) {
            int idx = tid + i * 256;
            int r   = idx >> 3;
            int c4  = (idx & 7) * 4;
            ra[i] = *(const v4f*)&A[(long long)(bm0 + r) * lda + k0 + c4];
        }
    };
    auto sstoreA = [&]() {
#pragma unroll
        for (int i = 0; i < 4; ++i) {
            int idx = tid + i * 256;
            int r   = idx >> 3;
            int c4  = (idx & 7) * 4;
            v4bf q;
            q[0] = (__bf16)ra[i][0]; q[1] = (__bf16)ra[i][1];
            q[2] = (__bf16)ra[i][2]; q[3] = (__bf16)ra[i][3];
            *(v4bf*)&As[r * LDS_A + c4] = q;
        }
    };
    // async DMA: global (bf16 W^T chunk) -> LDS, tracked with ASYNCcnt.
    auto asyncB = [&](int k0, int buf) {
        unsigned ldsoff =
            (unsigned)(unsigned long long)&Bs[buf * (BN * LDS_B) + b_n * LDS_B + b_c8];
        const __bf16* g = &Wt[(long long)(bn0 + b_n) * K + k0 + b_c8];
        asm volatile("global_load_async_to_lds_b128 %0, %1, off"
                     :: "v"(ldsoff), "v"(g)
                     : "memory");
    };

    v8f acc[2][2];
#pragma unroll
    for (int i = 0; i < 2; ++i)
#pragma unroll
        for (int j = 0; j < 2; ++j)
            acc[i][j] = (v8f){0.f, 0.f, 0.f, 0.f, 0.f, 0.f, 0.f, 0.f};

    const int frow = lane & 15;          // row (A) / col (B) within 16
    const int kb   = (lane >> 4) * 8;    // 0 or 8 : K sub-base per ISA layout

    gloadA(0);
    asyncB(0, 0);
    int buf = 0;
    for (int k0 = 0; k0 < K; k0 += BK) {
        sstoreA();
        // drain this wave's outstanding async-to-LDS before publishing tile
        asm volatile("s_wait_asynccnt 0x0" ::: "memory");
        __syncthreads();
        if (k0 + BK < K) {
            gloadA(k0 + BK);          // overlaps with WMMA below (LOADcnt)
            asyncB(k0 + BK, buf ^ 1); // overlaps with WMMA below (ASYNCcnt)
        }

        const int bofs = buf * (BN * LDS_B);
        BFrag af[2], bf[2];
#pragma unroll
        for (int i = 0; i < 2; ++i) {
            int r = wm * 32 + i * 16 + frow;
            af[i].h.lo = *(const v8bf*)&As[r * LDS_A + kb];
            af[i].h.hi = *(const v8bf*)&As[r * LDS_A + kb + 16];
        }
#pragma unroll
        for (int j = 0; j < 2; ++j) {
            int n = wn * 32 + j * 16 + frow;
            bf[j].h.lo = *(const v8bf*)&Bs[bofs + n * LDS_B + kb];
            bf[j].h.hi = *(const v8bf*)&Bs[bofs + n * LDS_B + kb + 16];
        }
#pragma unroll
        for (int i = 0; i < 2; ++i)
#pragma unroll
            for (int j = 0; j < 2; ++j)
                acc[i][j] = __builtin_amdgcn_wmma_f32_16x16x32_bf16(
                    false, af[i].v, false, bf[j].v,
                    (short)0, acc[i][j], false, false);
        __syncthreads();
        buf ^= 1;
    }

    // ---- epilogue + store (C/D layout: lane<16 -> M=r, else M=r+8) ---------
    const int mbase = (lane < 16) ? 0 : 8;
    const int ncol  = lane & 15;
#pragma unroll
    for (int i = 0; i < 2; ++i)
#pragma unroll
        for (int j = 0; j < 2; ++j) {
            int cg = bn0 + wn * 32 + j * 16 + ncol;
#pragma unroll
            for (int r = 0; r < 8; ++r) {
                int rg = bm0 + wm * 32 + i * 16 + mbase + r;
                float v = acc[i][j][r];
                if (EPI == 1)      v = softplusf(v + bias[cg]);
                else if (EPI == 2) v = geluf(v + bias[cg]);
                else if (EPI == 3) v += resp[(long long)rg * ldc + cg];
                else if (EPI == 4) v += bias[cg] + resp[(long long)rg * ldc + cg];
                C[(long long)rg * ldc + cg] = v;
            }
        }
}

// ------------------------- elementwise kernels ------------------------------
__global__ void k_diff(const float* __restrict__ x, float* __restrict__ xd,
                       long long total)
{
    long long idx = (long long)blockIdx.x * 256 + threadIdx.x;
    if (idx >= total) return;
    long long bl = idx / CD;
    int l = (int)(bl % CL);
    xd[idx] = (l == 0) ? x[idx] : x[idx] - x[idx - CD];
}

// W[K][N] f32 -> Wt[N][K] bf16
__global__ void k_wcvtT(const float* __restrict__ W, __bf16* __restrict__ Wt,
                        int K, int N)
{
    long long idx = (long long)blockIdx.x * 256 + threadIdx.x;
    if (idx >= (long long)K * N) return;
    int k = (int)(idx % K);
    int n = (int)(idx / K);
    Wt[idx] = (__bf16)W[(long long)k * N + n];
}

// causal depthwise conv1d over u-part of uz, then SiLU
__global__ void k_conv_silu(const float* __restrict__ uz,
                            const float* __restrict__ cw,
                            const float* __restrict__ cb,
                            float* __restrict__ u)
{
    long long idx = (long long)blockIdx.x * 256 + threadIdx.x;
    if (idx >= (long long)CB * CL * CDI) return;
    int d = (int)(idx & (CDI - 1));
    long long bl = idx >> 10;
    int l = (int)(bl % CL);
    long long b = bl / CL;
    float s = cb[d];
#pragma unroll
    for (int k = 0; k < 4; ++k) {
        int ls = l - 3 + k;
        if (ls >= 0)
            s += uz[(b * CL + ls) * (2 * CDI) + d] * cw[d * 4 + k];
    }
    u[idx] = siluf(s);
}

// fused selective scan + skip (u*Dp) + SiLU(z) gate
__global__ __launch_bounds__(256)
void k_scan(const float* __restrict__ u, const float* __restrict__ dt,
            const float* __restrict__ proj, const float* __restrict__ uz,
            const float* __restrict__ A_log, const float* __restrict__ Dp,
            float* __restrict__ y)
{
    __shared__ float Bsh[CDS];
    __shared__ float Csh[CDS];
    const int b    = blockIdx.x >> 2;          // 4 blocks per batch
    const int d    = (blockIdx.x & 3) * 256 + threadIdx.x;
    float A[CDS], h[CDS];
#pragma unroll
    for (int s = 0; s < CDS; ++s) {
        A[s] = -__expf(A_log[d * CDS + s]);
        h[s] = 0.f;
    }
    const float Dv = Dp[d];

    for (int l = 0; l < CL; ++l) {
        long long row = (long long)b * CL + l;
        if (threadIdx.x < 32) {
            int s = threadIdx.x & 15;
            float v = proj[row * CPROJ + CDTR + ((threadIdx.x < 16) ? 0 : CDS) + s];
            if (threadIdx.x < 16) Bsh[s] = v; else Csh[s] = v;
        }
        __syncthreads();
        float dtv = dt[row * CDI + d];
        float uv  = u[row * CDI + d];
        float accv = 0.f;
#pragma unroll
        for (int s = 0; s < CDS; ++s) {
            float dA = __expf(dtv * A[s]);
            h[s] = dA * h[s] + dtv * Bsh[s] * uv;
            accv += h[s] * Csh[s];
        }
        float zv = uz[row * (2 * CDI) + CDI + d];
        y[row * CDI + d] = (accv + uv * Dv) * siluf(zv);
        __syncthreads();
    }
}

__global__ __launch_bounds__(256)
void k_ln(const float* __restrict__ in, const float* __restrict__ g,
          const float* __restrict__ bta, float* __restrict__ out)
{
    __shared__ float red[256];
    const long long row = blockIdx.x;
    const float* p = in + row * CD;
    const int t = threadIdx.x;

    float s = p[t] + p[t + 256];
    red[t] = s; __syncthreads();
    for (int o = 128; o > 0; o >>= 1) {
        if (t < o) red[t] += red[t + o];
        __syncthreads();
    }
    float mu = red[0] / CD; __syncthreads();

    float a0 = p[t] - mu, a1 = p[t + 256] - mu;
    red[t] = a0 * a0 + a1 * a1; __syncthreads();
    for (int o = 128; o > 0; o >>= 1) {
        if (t < o) red[t] += red[t + o];
        __syncthreads();
    }
    float rstd = rsqrtf(red[0] / CD + 1e-5f);

    out[row * CD + t]       = a0 * rstd * g[t]       + bta[t];
    out[row * CD + t + 256] = a1 * rstd * g[t + 256] + bta[t + 256];
}

// adj = softmax(relu(e1 @ e2^T), axis=-1); one block per row
__global__ __launch_bounds__(256)
void k_adj(const float* __restrict__ e1, const float* __restrict__ e2,
           float* __restrict__ adj)
{
    __shared__ float red[256];
    const int n = blockIdx.x, t = threadIdx.x;
    float dot = 0.f;
#pragma unroll
    for (int k = 0; k < 16; ++k) dot += e1[n * 16 + k] * e2[t * 16 + k];
    dot = fmaxf(dot, 0.f);

    red[t] = dot; __syncthreads();
    for (int o = 128; o > 0; o >>= 1) {
        if (t < o) red[t] = fmaxf(red[t], red[t + o]);
        __syncthreads();
    }
    float mx = red[0]; __syncthreads();
    float e = __expf(dot - mx);
    red[t] = e; __syncthreads();
    for (int o = 128; o > 0; o >>= 1) {
        if (t < o) red[t] += red[t + o];
        __syncthreads();
    }
    adj[(long long)n * CL + t] = e / red[0];
}

// x1[b][l][d] f32 -> xT[b][d][l] bf16 (per-batch transposed operand for graph mix)
__global__ void k_xT(const float* __restrict__ x, __bf16* __restrict__ xT)
{
    long long idx = (long long)blockIdx.x * 256 + threadIdx.x;
    if (idx >= (long long)CB * CD * CL) return;
    int l = (int)(idx % CL);
    long long t = idx / CL;
    int d = (int)(t % CD);
    long long b = t / CD;
    xT[idx] = (__bf16)x[(b * CL + l) * CD + d];
}

// ------------------------------- host side ----------------------------------
static inline int cdiv_ll(long long a, int b) { return (int)((a + b - 1) / b); }

static void launch_gemm(hipStream_t st, int epi,
                        const float* A, int lda, long long bsA,
                        const __bf16* Wt, long long bsW,
                        float* C, int ldc, long long bsC,
                        const float* bias, const float* res, long long bsR,
                        int M, int N, int K, int batches)
{
    dim3 grid((N + BN - 1) / BN, (M + BM - 1) / BM, batches);
    dim3 blk(256);
    switch (epi) {
    case 0: k_gemm<0><<<grid, blk, 0, st>>>(A, lda, bsA, Wt, bsW, C, ldc, bsC, bias, res, bsR, M, N, K); break;
    case 1: k_gemm<1><<<grid, blk, 0, st>>>(A, lda, bsA, Wt, bsW, C, ldc, bsC, bias, res, bsR, M, N, K); break;
    case 2: k_gemm<2><<<grid, blk, 0, st>>>(A, lda, bsA, Wt, bsW, C, ldc, bsC, bias, res, bsR, M, N, K); break;
    case 3: k_gemm<3><<<grid, blk, 0, st>>>(A, lda, bsA, Wt, bsW, C, ldc, bsC, bias, res, bsR, M, N, K); break;
    case 4: k_gemm<4><<<grid, blk, 0, st>>>(A, lda, bsA, Wt, bsW, C, ldc, bsC, bias, res, bsR, M, N, K); break;
    }
}

extern "C" void kernel_launch(void* const* d_in, const int* in_sizes, int n_in,
                              void* d_out, int out_size, void* d_ws, size_t ws_size,
                              hipStream_t stream)
{
    (void)in_sizes; (void)n_in; (void)out_size; (void)ws_size;

    const float* x_in = (const float*)d_in[0];
    // d_ block params (indices 1..11), s_ block (12..22)
    const float* W_in[2]   = {(const float*)d_in[1],  (const float*)d_in[12]};
    const float* W_cw[2]   = {(const float*)d_in[2],  (const float*)d_in[13]};
    const float* W_cb[2]   = {(const float*)d_in[3],  (const float*)d_in[14]};
    const float* W_xp[2]   = {(const float*)d_in[4],  (const float*)d_in[15]};
    const float* W_dt[2]   = {(const float*)d_in[5],  (const float*)d_in[16]};
    const float* W_dtb[2]  = {(const float*)d_in[6],  (const float*)d_in[17]};
    const float* W_Al[2]   = {(const float*)d_in[7],  (const float*)d_in[18]};
    const float* W_Dp[2]   = {(const float*)d_in[8],  (const float*)d_in[19]};
    const float* W_out[2]  = {(const float*)d_in[9],  (const float*)d_in[20]};
    const float* W_lng[2]  = {(const float*)d_in[10], (const float*)d_in[21]};
    const float* W_lnb[2]  = {(const float*)d_in[11], (const float*)d_in[22]};
    const float* s_emb1 = (const float*)d_in[23];
    const float* s_emb2 = (const float*)d_in[24];
    const float* f_ln_g = (const float*)d_in[25];
    const float* f_ln_b = (const float*)d_in[26];
    const float* f_w1   = (const float*)d_in[27];
    const float* f_b1   = (const float*)d_in[28];
    const float* f_w2   = (const float*)d_in[29];
    const float* f_b2   = (const float*)d_in[30];

    float* out    = (float*)d_out;                       // x output (B*L*D)
    float* adjOut = out + (long long)CB * CL * CD;       // adj output (L*L)

    // ---- workspace bump allocation (deterministic layout) ----
    size_t off = 0;
    auto alloc = [&](size_t nbytes) -> void* {
        void* p = (char*)d_ws + off;
        off += (nbytes + 255) & ~(size_t)255;
        return p;
    };
    const long long NT = (long long)CB * CL;             // 8192 rows
    float* xd   = (float*)alloc(NT * CD * 4);
    float* uz   = (float*)alloc(NT * 2 * CDI * 4);
    float* ub   = (float*)alloc(NT * CDI * 4);
    float* proj = (float*)alloc(NT * CPROJ * 4);
    float* dtb  = (float*)alloc(NT * CDI * 4);
    float* yb   = (float*)alloc(NT * CDI * 4);
    float* pre  = (float*)alloc(NT * CD * 4);
    float* x1   = (float*)alloc(NT * CD * 4);
    float* xg   = (float*)alloc(NT * CD * 4);
    float* x2   = (float*)alloc(NT * CD * 4);
    float* hln  = (float*)alloc(NT * CD * 4);
    __bf16* T_in[2], *T_xp[2], *T_dt[2], *T_out[2];
    for (int p = 0; p < 2; ++p) {
        T_in[p]  = (__bf16*)alloc((size_t)(2 * CDI) * CD * 2);   // 2048x512
        T_xp[p]  = (__bf16*)alloc((size_t)CPROJ * CDI * 2);      // 64x1024
        T_dt[p]  = (__bf16*)alloc((size_t)CDI * CDTR * 2);       // 1024x32
        T_out[p] = (__bf16*)alloc((size_t)CD * CDI * 2);         // 512x1024
    }
    __bf16* T_fw1 = (__bf16*)alloc((size_t)CDFF * CD * 2);       // 2048x512
    __bf16* T_fw2 = (__bf16*)alloc((size_t)CD * CDFF * 2);       // 512x2048
    __bf16* xT    = (__bf16*)alloc((size_t)CB * CD * CL * 2);    // per-batch x^T

    // ---- weight transpose+convert to bf16 ----
    auto wcvt = [&](const float* W, __bf16* Wt, int K, int N) {
        long long tot = (long long)K * N;
        k_wcvtT<<<cdiv_ll(tot, 256), 256, 0, stream>>>(W, Wt, K, N);
    };
    for (int p = 0; p < 2; ++p) {
        wcvt(W_in[p],  T_in[p],  CD,   2 * CDI);
        wcvt(W_xp[p],  T_xp[p],  CDI,  CPROJ);
        wcvt(W_dt[p],  T_dt[p],  CDTR, CDI);
        wcvt(W_out[p], T_out[p], CDI,  CD);
    }
    wcvt(f_w1, T_fw1, CD,   CDFF);
    wcvt(f_w2, T_fw2, CDFF, CD);

    // ---- Mamba block runner: mamba_in -> pre = res + mamba(mamba_in); LN -> xout
    auto mamba_block = [&](int p, const float* mamba_in, const float* res,
                           float* xout) {
        // in_proj: uz = mamba_in @ in_w    (8192 x 2048, K=512)
        launch_gemm(stream, 0, mamba_in, CD, 0, T_in[p], 0,
                    uz, 2 * CDI, 0, nullptr, nullptr, 0,
                    (int)NT, 2 * CDI, CD, 1);
        // causal depthwise conv + SiLU on u-half
        k_conv_silu<<<cdiv_ll(NT * CDI, 256), 256, 0, stream>>>(
            uz, W_cw[p], W_cb[p], ub);
        // x_proj: proj = u @ xproj_w       (8192 x 64, K=1024)
        launch_gemm(stream, 0, ub, CDI, 0, T_xp[p], 0,
                    proj, CPROJ, 0, nullptr, nullptr, 0,
                    (int)NT, CPROJ, CDI, 1);
        // dt = softplus(proj[:, :32] @ dt_w + dt_b)  (8192 x 1024, K=32)
        launch_gemm(stream, 1, proj, CPROJ, 0, T_dt[p], 0,
                    dtb, CDI, 0, W_dtb[p], nullptr, 0,
                    (int)NT, CDI, CDTR, 1);
        // selective scan + skip + gate
        k_scan<<<CB * (CDI / 256), 256, 0, stream>>>(
            ub, dtb, proj, uz, W_Al[p], W_Dp[p], yb);
        // out_proj + residual: pre = res + y @ out_w  (8192 x 512, K=1024)
        launch_gemm(stream, 3, yb, CDI, 0, T_out[p], 0,
                    pre, CD, 0, nullptr, res, 0,
                    (int)NT, CD, CDI, 1);
        // layernorm
        k_ln<<<(int)NT, 256, 0, stream>>>(pre, W_lng[p], W_lnb[p], xout);
    };

    // ================= block 1: DiffSSM =================
    k_diff<<<cdiv_ll(NT * CD, 256), 256, 0, stream>>>(x_in, xd, NT * CD);
    mamba_block(0, xd, x_in, x1);

    // ================= graph mix =================
    k_adj<<<CL, 256, 0, stream>>>(s_emb1, s_emb2, adjOut);
    k_xT<<<cdiv_ll((long long)CB * CD * CL, 256), 256, 0, stream>>>(x1, xT);
    // xg[b] = adj (256x256) @ x1[b] (256x512)   — batched WMMA GEMM
    launch_gemm(stream, 0, adjOut, CL, 0,
                xT, (long long)CD * CL,
                xg, CD, (long long)CL * CD,
                nullptr, nullptr, 0,
                CL, CD, CL, CB);

    // ================= block 2: SpatialGraphMamba =================
    mamba_block(1, xg, x1, x2);

    // ================= FFN =================
    k_ln<<<(int)NT, 256, 0, stream>>>(x2, f_ln_g, f_ln_b, hln);
    // h1 = gelu(hln @ f_w1 + f_b1)   (8192 x 2048, K=512) -> reuse uz
    launch_gemm(stream, 2, hln, CD, 0, T_fw1, 0,
                uz, CDFF, 0, f_b1, nullptr, 0,
                (int)NT, CDFF, CD, 1);
    // out = x2 + h1 @ f_w2 + f_b2    (8192 x 512, K=2048)
    launch_gemm(stream, 4, uz, CDFF, 0, T_fw2, 0,
                out, CD, 0, f_b2, x2, 0,
                (int)NT, CD, CDFF, 1);
}
